// HumanVAttention_45414984188591
// MI455X (gfx1250) — matrix operand order repaired and checked by reading, not compiled, not run on hardware
//
#include <hip/hip_runtime.h>
#include <hip/hip_bf16.h>

#define B_     2
#define T_     4096
#define HID_   2048
#define NH_    16
#define KVH_   4
#define HD_    128
#define NB_    64            // T_/64 query/key blocks
#define QDIM_  (NH_*HD_)     // 2048
#define KVDIM_ (KVH_*HD_)    // 512
#define BT_    (B_*T_)       // 8192

typedef __bf16 bf16_t;
typedef __attribute__((ext_vector_type(16))) __bf16 v16b;
typedef __attribute__((ext_vector_type(8)))  __bf16 v8b;
typedef __attribute__((ext_vector_type(8)))  float  v8f;
typedef __attribute__((ext_vector_type(4)))  unsigned int v4u;
typedef __attribute__((ext_vector_type(8)))  int v8i;
typedef __attribute__((ext_vector_type(4)))  int v4i;

static __device__ inline v16b join8(v8b lo, v8b hi) {
  v16b r;
#pragma unroll
  for (int i = 0; i < 8; ++i) { r[i] = lo[i]; r[i + 8] = hi[i]; }
  return r;
}
static __device__ inline float half_max(float v) {
#pragma unroll
  for (int off = 8; off > 0; off >>= 1) v = fmaxf(v, __shfl_xor(v, off, 32));
  return v;  // reduced over 16-lane halves (rows of C/D fragments)
}
static __device__ inline float half_sum(float v) {
#pragma unroll
  for (int off = 8; off > 0; off >>= 1) v += __shfl_xor(v, off, 32);
  return v;
}
// LDS aperture flat address -> 32-bit LDS byte offset (ISA 10.2: addr[31:0])
static __device__ inline unsigned lds_off(const void* p) {
  return (unsigned)(unsigned long long)(uintptr_t)p;
}

// ---- Tensor Data Mover: 2-D tile load, bf16 elements (data_size code 1) ----
// Packs the tile row-major into LDS (Laddr advances contiguously per D# spec).
// This toolchain's builtin takes 6 args (g0,g1,g2,g3,g4,cpol).
static __device__ inline void tdm_load_2d_bf16(unsigned ldsByteOff, const void* gsrc,
                                               unsigned tile_d0, unsigned tile_d1,
                                               unsigned long long stride0_elems) {
  unsigned long long ga = (unsigned long long)(uintptr_t)gsrc;
  v4u g0;
  g0[0] = 1u;                                             // count=1, user D#
  g0[1] = ldsByteOff;                                     // lds_addr (bytes)
  g0[2] = (unsigned)(ga & 0xFFFFFFFFu);                   // global_addr[31:0]
  g0[3] = (unsigned)((ga >> 32) & 0x01FFFFFFu) | (2u << 30); // addr[56:32], type=2
  v8i g1;
  unsigned td0 = tile_d0, td1 = tile_d1;                  // tensor_dim == tile_dim
  g1[0] = (int)(1u << 16);                                // data_size=2B; mask=0
  g1[1] = (int)((td0 & 0xFFFFu) << 16);                   // tensor_dim0 lo16
  g1[2] = (int)((td0 >> 16) | ((td1 & 0xFFFFu) << 16));   // dim0 hi / dim1 lo
  g1[3] = (int)((td1 >> 16) | (tile_d0 << 16));           // dim1 hi / tile_dim0
  g1[4] = (int)(tile_d1);                                 // tile_dim1; tile_dim2=0
  g1[5] = (int)(stride0_elems & 0xFFFFFFFFull);           // tensor_dim0_stride lo
  g1[6] = (int)((unsigned)(stride0_elems >> 32) & 0xFFFFu); // stride0 hi; stride1=0
  g1[7] = 0;
  v4i gz4 = {0, 0, 0, 0};
  v8i gz8 = {0, 0, 0, 0, 0, 0, 0, 0};
  __builtin_amdgcn_tensor_load_to_lds(g0, g1, gz4, gz4, gz8, 0);
}

// ---------------- fp32 -> bf16 cast ----------------
__global__ void hva_cast_bf16(const float* __restrict__ in, bf16_t* __restrict__ out, int n) {
  int i = (blockIdx.x * blockDim.x + threadIdx.x) * 4;
  if (i + 3 < n) {
#pragma unroll
    for (int j = 0; j < 4; ++j) out[i + j] = (bf16_t)in[i + j];
  }
}

// ---------------- bf16 WMMA GEMM: C[M,N] = A[M,K] * W[K,N] ----------------
// 256 threads = 8 waves; workgroup tile 128x128; wave tile 32x64 (2x4 frags);
// BK=32; TDM double-buffered LDS staging (wave 0 drives the DMA).
template <bool OUT_F32>
__global__ __launch_bounds__(256)
void hva_gemm_bf16(const bf16_t* __restrict__ A, const bf16_t* __restrict__ W,
                   void* __restrict__ Cout, int M, int N, int K) {
  __shared__ bf16_t Asm[2][128 * 32];
  __shared__ bf16_t Wsm[2][32 * 128];
  const int tid  = threadIdx.x;
  const int wave = tid >> 5;
  const int lane = tid & 31;
  const int wm   = wave >> 1;        // 0..3  (M direction)
  const int wn   = wave & 1;         // 0..1  (N direction)
  const int m0   = blockIdx.y * 128;
  const int n0   = blockIdx.x * 128;

  v8f acc[2][4];
#pragma unroll
  for (int mt = 0; mt < 2; ++mt)
#pragma unroll
    for (int nt = 0; nt < 4; ++nt)
#pragma unroll
      for (int r = 0; r < 8; ++r) acc[mt][nt][r] = 0.0f;

  const int arow = lane & 15;
  const int akb  = (lane >> 4) * 8;
  const int nIter = K >> 5;

  if (wave == 0) {  // prologue: stage 0 into buffer 0
    tdm_load_2d_bf16(lds_off(&Asm[0][0]), A + (size_t)m0 * K, 32, 128, (unsigned long long)K);
    tdm_load_2d_bf16(lds_off(&Wsm[0][0]), W + n0,            128, 32, (unsigned long long)N);
  }

  for (int it = 0; it < nIter; ++it) {
    const int cur = it & 1;
    if (wave == 0) {
      if (it + 1 < nIter) {  // stage it+1 into the other buffer
        int k1 = (it + 1) << 5;
        tdm_load_2d_bf16(lds_off(&Asm[cur ^ 1][0]), A + (size_t)m0 * K + k1,
                         32, 128, (unsigned long long)K);
        tdm_load_2d_bf16(lds_off(&Wsm[cur ^ 1][0]), W + (size_t)k1 * N + n0,
                         128, 32, (unsigned long long)N);
        __builtin_amdgcn_s_wait_tensorcnt(2);  // in-order: stage `it` complete
      } else {
        __builtin_amdgcn_s_wait_tensorcnt(0);
      }
    }
    __syncthreads();  // buffer `cur` ready for all waves

    const bf16_t* As = &Asm[cur][0];
    const bf16_t* Ws = &Wsm[cur][0];
    v16b afr[2], bfr[4];
#pragma unroll
    for (int mt = 0; mt < 2; ++mt) {
      int r = wm * 32 + mt * 16 + arow;
      afr[mt] = join8(*(const v8b*)&As[r * 32 + akb],
                      *(const v8b*)&As[r * 32 + akb + 16]);
    }
#pragma unroll
    for (int nt = 0; nt < 4; ++nt)
      for (int s = 0; s < 16; ++s)
        bfr[nt][s] = Ws[(akb + (s & 7) + ((s >> 3) << 4)) * 128 + wn * 64 + nt * 16 + arow];

#pragma unroll
    for (int mt = 0; mt < 2; ++mt)
#pragma unroll
      for (int nt = 0; nt < 4; ++nt)
        acc[mt][nt] = __builtin_amdgcn_wmma_f32_16x16x32_bf16(
            false, afr[mt], false, bfr[nt], (short)0, acc[mt][nt], false, false);
    __syncthreads();  // all waves done reading buffer `cur`
  }

#pragma unroll
  for (int mt = 0; mt < 2; ++mt)
#pragma unroll
    for (int nt = 0; nt < 4; ++nt)
#pragma unroll
      for (int r = 0; r < 8; ++r) {
        int row = m0 + wm * 32 + mt * 16 + r + 8 * (lane >> 4);
        int col = n0 + wn * 64 + nt * 16 + (lane & 15);
        if (OUT_F32) ((float*)Cout)[(size_t)row * N + col] = acc[mt][nt][r];
        else         ((bf16_t*)Cout)[(size_t)row * N + col] = (bf16_t)acc[mt][nt][r];
      }
}

// ---------------- RoPE + repack ----------------
// Qr[b][h][t][d]    = rope(Q)*HD^-0.5 (pre-scaled)
// KtT[b][kvh][d][t] = rope(K) transposed (key index contiguous per d)
// Vr[b][kvh][t][d]  = V
__global__ __launch_bounds__(128)
void hva_rope_pack(const bf16_t* __restrict__ Qt, const bf16_t* __restrict__ Kt,
                   const bf16_t* __restrict__ Vt, const float* __restrict__ cosp,
                   const float* __restrict__ sinp, bf16_t* __restrict__ Qr,
                   bf16_t* __restrict__ KtT, bf16_t* __restrict__ Vr) {
  int bt = blockIdx.x;                 // 0..B*T-1
  int b  = bt / T_, t = bt % T_;
  int d  = threadIdx.x;                // 0..127
  float c = cosp[(size_t)bt * HD_ + d];
  float s = sinp[(size_t)bt * HD_ + d];
  const float qscale = 0.08838834764831845f;  // 128^-0.5
#pragma unroll
  for (int h = 0; h < NH_; ++h) {
    const bf16_t* qp = Qt + (size_t)bt * QDIM_ + h * HD_;
    float x  = (float)qp[d];
    float xr = (d < 64) ? -(float)qp[d + 64] : (float)qp[d - 64];
    Qr[(((size_t)b * NH_ + h) * T_ + t) * HD_ + d] = (bf16_t)((x * c + xr * s) * qscale);
  }
#pragma unroll
  for (int kh = 0; kh < KVH_; ++kh) {
    const bf16_t* kp = Kt + (size_t)bt * KVDIM_ + kh * HD_;
    float x  = (float)kp[d];
    float xr = (d < 64) ? -(float)kp[d + 64] : (float)kp[d - 64];
    KtT[(((size_t)b * KVH_ + kh) * HD_ + d) * T_ + t] = (bf16_t)(x * c + xr * s);
    Vr[(((size_t)b * KVH_ + kh) * T_ + t) * HD_ + d] =
        Vt[(size_t)bt * KVDIM_ + kh * HD_ + d];
  }
}

// ---------------- block-sparse flash attention ----------------
// 1 workgroup per (b, h, qblock); 4 waves x 16 query rows; online softmax;
// K/V tiles DMA'd by the Tensor Data Mover (wave 0 issues, TENSORcnt sync).
__global__ __launch_bounds__(128)
void hva_attn(const bf16_t* __restrict__ Qr, const bf16_t* __restrict__ KtT,
              const bf16_t* __restrict__ Vr, const int* __restrict__ amask,
              bf16_t* __restrict__ AttnOut) {
  __shared__ bf16_t Klds[128 * 64];      // [d][key]
  __shared__ bf16_t Vlds[64 * 128];      // [key][d]
  __shared__ bf16_t Plds[4 * 16 * 64];   // per-wave probs [row][key]

  int bid = blockIdx.x;
  int qb  = bid % NB_;
  int h   = (bid / NB_) % NH_;
  int b   = bid / (NB_ * NH_);
  int kvh = h / (NH_ / KVH_);
  int tid = threadIdx.x, wave = tid >> 5, lane = tid & 31;

  // --- selected key blocks (matches reference _select_key_blocks) ---
  unsigned long long selm = 1ull;                       // block 0 always
  int lo = qb - 3; if (lo < 0) lo = 0;
  for (int j = lo; j <= qb; ++j) selm |= 1ull << j;     // local window
  int gcount = qb / 4 + 1;                              // multiples of STRIDE
  int gstart = gcount > 2 ? gcount - 2 : 0;             // keep last GLOBAL_NB
  for (int g = gstart; g < gcount; ++g) selm |= 1ull << (g * 4);
  int sel[8], ns = 0;
  for (int j = 0; j <= qb; ++j)
    if ((selm >> j) & 1ull) sel[ns++] = j;

  // --- Q fragments (held in registers for whole kernel) ---
  const int qrow = qb * 64 + wave * 16 + (lane & 15);
  const bf16_t* qp = Qr + (((size_t)b * NH_ + h) * T_ + qrow) * HD_;
  const int akb = (lane >> 4) * 8;
  v16b qfr[4];
#pragma unroll
  for (int kc = 0; kc < 4; ++kc)
    qfr[kc] = join8(*(const v8b*)&qp[kc * 32 + akb],
                    *(const v8b*)&qp[kc * 32 + akb + 16]);

  float row_m[8], row_l[8];
  v8f oacc[8];
#pragma unroll
  for (int r = 0; r < 8; ++r) { row_m[r] = -1e30f; row_l[r] = 0.0f; }
#pragma unroll
  for (int dt = 0; dt < 8; ++dt)
#pragma unroll
    for (int r = 0; r < 8; ++r) oacc[dt][r] = 0.0f;

  const bf16_t* Ktb = KtT + ((size_t)b * KVH_ + kvh) * (size_t)HD_ * T_;
  const bf16_t* Vb  = Vr  + ((size_t)b * KVH_ + kvh) * (size_t)T_ * HD_;
  bf16_t* Pw = &Plds[wave * 16 * 64];

  for (int s = 0; s < ns; ++s) {
    int j  = sel[s];
    int kb = j * 64;
    __syncthreads();  // previous iteration done reading K/V LDS
    if (wave == 0) {
      // K^T tile: 128 d-rows x 64 keys, row stride T_   -> Klds[128][64]
      tdm_load_2d_bf16(lds_off(&Klds[0]), Ktb + kb, 64, 128, (unsigned long long)T_);
      // V tile: 64 key-rows x 128 d, row stride HD_     -> Vlds[64][128]
      tdm_load_2d_bf16(lds_off(&Vlds[0]), Vb + (size_t)kb * HD_, 128, 64,
                       (unsigned long long)HD_);
      __builtin_amdgcn_s_wait_tensorcnt(0);
    }
    __syncthreads();

    // ---- scores: Q(16x128) x K^T(128x64) -> 4 N-tiles ----
    v8f sfr[4];
#pragma unroll
    for (int nt = 0; nt < 4; ++nt) {
      v8f sc;
#pragma unroll
      for (int r = 0; r < 8; ++r) sc[r] = 0.0f;
#pragma unroll
      for (int kc = 0; kc < 4; ++kc) {
        v16b bfr;
        for (int s = 0; s < 16; ++s)
          bfr[s] = Klds[(kc * 32 + akb + (s & 7) + ((s >> 3) << 4)) * 64 + nt * 16 + (lane & 15)];
        sc = __builtin_amdgcn_wmma_f32_16x16x32_bf16(false, qfr[kc], false, bfr,
                                                     (short)0, sc, false, false);
      }
      sfr[nt] = sc;
    }

    // ---- masks + online softmax (per 16-lane half == per query row) ----
    float kvbias[4];
#pragma unroll
    for (int nt = 0; nt < 4; ++nt)
      kvbias[nt] = amask[b * T_ + kb + nt * 16 + (lane & 15)] ? 0.0f : -1e9f;
    bool selfblk = (j == qb);

#pragma unroll
    for (int r = 0; r < 8; ++r) {
      int qloc = wave * 16 + r + 8 * (lane >> 4);   // query row within 64-block
      float mx = -1e30f;
#pragma unroll
      for (int nt = 0; nt < 4; ++nt) {
        int kloc = nt * 16 + (lane & 15);
        float v = sfr[nt][r] + kvbias[nt];
        if (selfblk && kloc > qloc) v += -1e9f;     // causal triangle in self block
        sfr[nt][r] = v;
        mx = fmaxf(mx, v);
      }
      float bmax = half_max(mx);
      float newm = fmaxf(row_m[r], bmax);
      float scl  = __expf(row_m[r] - newm);
      float psum = 0.0f;
#pragma unroll
      for (int nt = 0; nt < 4; ++nt) {
        float p = __expf(sfr[nt][r] - newm);
        sfr[nt][r] = p;
        psum += p;
      }
      psum = half_sum(psum);
      row_l[r] = row_l[r] * scl + psum;
      row_m[r] = newm;
#pragma unroll
      for (int dt = 0; dt < 8; ++dt) oacc[dt][r] *= scl;
    }

    // ---- re-layout P (C/D frag -> A frag) through per-wave LDS ----
#pragma unroll
    for (int nt = 0; nt < 4; ++nt)
#pragma unroll
      for (int r = 0; r < 8; ++r)
        Pw[(r + 8 * (lane >> 4)) * 64 + nt * 16 + (lane & 15)] = (bf16_t)sfr[nt][r];
    // same-wave DS ordering; Pw is private to this wave

    v16b pfr[2];
#pragma unroll
    for (int kc = 0; kc < 2; ++kc) {
      int pb = kc * 32 + akb;
      pfr[kc] = join8(*(const v8b*)&Pw[(lane & 15) * 64 + pb],
                      *(const v8b*)&Pw[(lane & 15) * 64 + pb + 16]);
    }
#pragma unroll
    for (int dt = 0; dt < 8; ++dt)
#pragma unroll
      for (int kc = 0; kc < 2; ++kc) {
        v16b vfr;
        for (int s = 0; s < 16; ++s)
          vfr[s] = Vlds[(kc * 32 + akb + (s & 7) + ((s >> 3) << 4)) * 128 + dt * 16 + (lane & 15)];
        oacc[dt] = __builtin_amdgcn_wmma_f32_16x16x32_bf16(false, pfr[kc], false, vfr,
                                                           (short)0, oacc[dt], false, false);
      }
  }

  // ---- normalize + store [B][T][NH*HD] bf16 ----
#pragma unroll
  for (int dt = 0; dt < 8; ++dt)
#pragma unroll
    for (int r = 0; r < 8; ++r) {
      float v = oacc[dt][r] / row_l[r];
      int trow = qb * 64 + wave * 16 + r + 8 * (lane >> 4);
      AttnOut[((size_t)b * T_ + trow) * QDIM_ + h * HD_ + dt * 16 + (lane & 15)] = (bf16_t)v;
    }
}

// ---------------- host launch ----------------
extern "C" void kernel_launch(void* const* d_in, const int* in_sizes, int n_in,
                              void* d_out, int out_size, void* d_ws, size_t ws_size,
                              hipStream_t stream) {
  const float* hs   = (const float*)d_in[0];
  const float* cosp = (const float*)d_in[1];
  const float* sinp = (const float*)d_in[2];
  const float* Wq   = (const float*)d_in[3];
  const float* Wk   = (const float*)d_in[4];
  const float* Wv   = (const float*)d_in[5];
  const float* Wo   = (const float*)d_in[6];
  const int*   am   = (const int*)d_in[7];
  float* out = (float*)d_out;

  char* ws = (char*)d_ws;
  size_t off = 0;
  auto carve = [&](size_t bytes) -> void* {
    void* p = ws + off;
    off += (bytes + 255) & ~(size_t)255;
    return p;
  };
  bf16_t* Xb   = (bf16_t*)carve((size_t)BT_ * HID_ * 2);     // 32 MB (reused as Qr)
  bf16_t* Wqb  = (bf16_t*)carve((size_t)HID_ * QDIM_ * 2);   // 8 MB
  bf16_t* Wkb  = (bf16_t*)carve((size_t)HID_ * KVDIM_ * 2);  // 2 MB
  bf16_t* Wvb  = (bf16_t*)carve((size_t)HID_ * KVDIM_ * 2);  // 2 MB
  bf16_t* Wob  = (bf16_t*)carve((size_t)QDIM_ * HID_ * 2);   // 8 MB
  bf16_t* Qtmp = (bf16_t*)carve((size_t)BT_ * QDIM_ * 2);    // 32 MB (reused as AttnOut)
  bf16_t* Ktmp = (bf16_t*)carve((size_t)BT_ * KVDIM_ * 2);   // 8 MB
  bf16_t* Vtmp = (bf16_t*)carve((size_t)BT_ * KVDIM_ * 2);   // 8 MB
  bf16_t* KtT  = (bf16_t*)carve((size_t)BT_ * KVDIM_ * 2);   // 8 MB
  bf16_t* Vr   = (bf16_t*)carve((size_t)BT_ * KVDIM_ * 2);   // 8 MB
  bf16_t* Qr      = Xb;    // X dead after projections
  bf16_t* AttnOut = Qtmp;  // Qtmp dead after rope_pack

  // 1) casts
  hva_cast_bf16<<<(BT_ * HID_) / 1024, 256, 0, stream>>>(hs, Xb, BT_ * HID_);
  hva_cast_bf16<<<(HID_ * QDIM_) / 1024, 256, 0, stream>>>(Wq, Wqb, HID_ * QDIM_);
  hva_cast_bf16<<<(HID_ * KVDIM_) / 1024, 256, 0, stream>>>(Wk, Wkb, HID_ * KVDIM_);
  hva_cast_bf16<<<(HID_ * KVDIM_) / 1024, 256, 0, stream>>>(Wv, Wvb, HID_ * KVDIM_);
  hva_cast_bf16<<<(QDIM_ * HID_) / 1024, 256, 0, stream>>>(Wo, Wob, QDIM_ * HID_);

  // 2) projections (bf16 WMMA, TDM-staged)
  hva_gemm_bf16<false><<<dim3(QDIM_ / 128, BT_ / 128), 256, 0, stream>>>(
      Xb, Wqb, Qtmp, BT_, QDIM_, HID_);
  hva_gemm_bf16<false><<<dim3(KVDIM_ / 128, BT_ / 128), 256, 0, stream>>>(
      Xb, Wkb, Ktmp, BT_, KVDIM_, HID_);
  hva_gemm_bf16<false><<<dim3(KVDIM_ / 128, BT_ / 128), 256, 0, stream>>>(
      Xb, Wvb, Vtmp, BT_, KVDIM_, HID_);

  // 3) rope + repack (writes Qr over Xb)
  hva_rope_pack<<<BT_, 128, 0, stream>>>(Qtmp, Ktmp, Vtmp, cosp, sinp, Qr, KtT, Vr);

  // 4) block-sparse flash attention (writes AttnOut over Qtmp)
  hva_attn<<<B_ * NH_ * NB_, 128, 0, stream>>>(Qr, KtT, Vr, am, AttnOut);

  // 5) output projection -> fp32 d_out
  hva_gemm_bf16<true><<<dim3(HID_ / 128, BT_ / 128), 256, 0, stream>>>(
      AttnOut, Wob, out, BT_, HID_, QDIM_);
}
